// AttentionBlock_20306605375962
// MI455X (gfx1250) — compile-verified, matrix operand order checked by reading, not run on hardware
//
#include <hip/hip_runtime.h>
#include <hip/hip_bf16.h>

typedef __attribute__((ext_vector_type(16))) __bf16 v16bf;
typedef __attribute__((ext_vector_type(8)))  __bf16 v8bf;
typedef __attribute__((ext_vector_type(8)))  float  v8f;

#define B_  8
#define C_  512
#define T_  2048
#define H_  4
#define CH_ 128
#define O3_ 1536

__device__ __forceinline__ __bf16 f2bf(float f) {
  unsigned u = __builtin_bit_cast(unsigned, f);
  unsigned r = u + 0x7FFFu + ((u >> 16) & 1u);
  unsigned short h = (unsigned short)(r >> 16);
  return __builtin_bit_cast(__bf16, h);
}

// Load a 16x32-bf16 WMMA A/B fragment row: per lane, 16 values of one
// row/column split as K = {koff..koff+7, 16+koff..23+koff}, koff = 8*(lane/16).
// Two contiguous 16-byte loads.
__device__ __forceinline__ v16bf load_frag(const __bf16* p, int koff) {
  v8bf lo = *(const v8bf*)(p + koff);
  v8bf hi = *(const v8bf*)(p + 16 + koff);
  v16bf r;
#pragma unroll
  for (int i = 0; i < 8; ++i) { r[i] = lo[i]; r[i + 8] = hi[i]; }
  return r;
}

__device__ __forceinline__ v8f wmma_bf16(v16bf a, v16bf b, v8f c) {
  return __builtin_amdgcn_wmma_f32_16x16x32_bf16(false, a, false, b, (short)0, c,
                                                 false, false);
}

// ds_swizzle XOR reductions confined to 16-lane halves (xor masks <= 8)
#define SWZ(v, patt) \
  __builtin_bit_cast(float, __builtin_amdgcn_ds_swizzle(__builtin_bit_cast(int, (v)), (patt)))

// ---------------------------------------------------------------- converters
__global__ void k_convert(const float* __restrict__ src, __bf16* __restrict__ dst, int n) {
  int i = blockIdx.x * blockDim.x + threadIdx.x;
  if (i < n) dst[i] = f2bf(src[i]);
}

// x (b, c, T) fp32 -> xT (b, T, c) bf16 via LDS tile
__global__ void k_transpose_x(const float* __restrict__ x, __bf16* __restrict__ xT) {
  __shared__ float tile[32][33];
  int b = blockIdx.z;
  int c0 = blockIdx.y * 32, t0 = blockIdx.x * 32;
  int tx = threadIdx.x & 31, ty = threadIdx.x >> 5;  // 32 x 8
#pragma unroll
  for (int i = 0; i < 32; i += 8)
    tile[ty + i][tx] = x[((size_t)b * C_ + c0 + ty + i) * T_ + t0 + tx];
  __syncthreads();
#pragma unroll
  for (int i = 0; i < 32; i += 8)
    xT[((size_t)b * T_ + t0 + ty + i) * C_ + c0 + tx] = f2bf(tile[tx][ty + i]);
}

// ---------------------------------------------------------------- pipelined GEMM core
// acc[4] += W[mrow, :] x Xrows[nbase.., :]; software-pipelined: next K-step's
// fragments load into fresh registers while current WMMAs execute.
__device__ __forceinline__ void gemm_k512(v8f acc[4], const __bf16* __restrict__ arow,
                                          const __bf16* __restrict__ bbase, int koff) {
  v16bf a = load_frag(arow, koff);
  v16bf b[4];
#pragma unroll
  for (int j = 0; j < 4; ++j) b[j] = load_frag(bbase + (size_t)j * 16 * C_, koff);

  for (int kb = 0; kb < C_ - 32; kb += 32) {
    v16bf an = load_frag(arow + kb + 32, koff);
    v16bf bn[4];
#pragma unroll
    for (int j = 0; j < 4; ++j)
      bn[j] = load_frag(bbase + (size_t)j * 16 * C_ + kb + 32, koff);
    __builtin_prefetch(arow + kb + 64, 0, 3);
#pragma unroll
    for (int j = 0; j < 4; ++j) acc[j] = wmma_bf16(a, b[j], acc[j]);
    a = an;
#pragma unroll
    for (int j = 0; j < 4; ++j) b[j] = bn[j];
  }
#pragma unroll
  for (int j = 0; j < 4; ++j) acc[j] = wmma_bf16(a, b[j], acc[j]);
}

// ---------------------------------------------------------------- QKV GEMM
__global__ void k_qkv(const __bf16* __restrict__ wq, const __bf16* __restrict__ xT,
                      const float* __restrict__ bias,
                      __bf16* __restrict__ Q, __bf16* __restrict__ Kb,
                      __bf16* __restrict__ Vt) {
  int b = blockIdx.z;
  int w = threadIdx.x >> 5, lane = threadIdx.x & 31;
  int hi = lane >> 4, ln = lane & 15;
  int koff = hi * 8;
  int mbase = blockIdx.y * 64 + w * 16;
  int nbase = blockIdx.x * 64;
  v8f acc[4] = {};
  const __bf16* arow = wq + (size_t)(mbase + ln) * C_;
  const __bf16* bbase = xT + ((size_t)b * T_ + nbase + ln) * C_;
  gemm_k512(acc, arow, bbase, koff);

  const float SCALEQ = 0.08838834764831845f;  // 1/sqrt(128): full softmax scale on Q
#pragma unroll
  for (int j = 0; j < 4; ++j) {
    int t = nbase + j * 16 + ln;
#pragma unroll
    for (int r = 0; r < 8; ++r) {
      int o = mbase + r + hi * 8;
      float v = acc[j][r] + bias[o];
      if (o < C_) {
        int h = o >> 7, ch = o & 127;
        Q[(((size_t)b * H_ + h) * T_ + t) * CH_ + ch] = f2bf(v * SCALEQ);
      } else if (o < 2 * C_) {
        int o2 = o - C_;
        int h = o2 >> 7, ch = o2 & 127;
        Kb[(((size_t)b * H_ + h) * T_ + t) * CH_ + ch] = f2bf(v);
      } else {
        int o2 = o - 2 * C_;
        int h = o2 >> 7, ch = o2 & 127;
        Vt[(((size_t)b * H_ + h) * CH_ + ch) * T_ + t] = f2bf(v);
      }
    }
  }
}

// ---------------------------------------------------------------- flash attention
__global__ void k_attn(const __bf16* __restrict__ Q, const __bf16* __restrict__ Kb,
                       const __bf16* __restrict__ Vt, __bf16* __restrict__ Ab) {
  __shared__ __bf16 plds[4][16][32];
  int bh = blockIdx.y;
  int w = threadIdx.x >> 5, lane = threadIdx.x & 31;
  int hi = lane >> 4, ln = lane & 15;
  int koff = hi * 8;
  int qbase = blockIdx.x * 64 + w * 16;

  // pin Q fragments (16 rows x 128 ch = 4 K-steps) in VGPRs
  v16bf qfrag[4];
  const __bf16* qrow = Q + ((size_t)bh * T_ + qbase + ln) * CH_;
#pragma unroll
  for (int kk = 0; kk < 4; ++kk) qfrag[kk] = load_frag(qrow + kk * 32, koff);

  v8f oacc[8] = {};          // 16 rows x 128 ch output accumulators
  float mrow[8], lrow[8];    // per-row running max / sum (row = r + 8*hi)
#pragma unroll
  for (int r = 0; r < 8; ++r) { mrow[r] = -3.0e38f; lrow[r] = 0.f; }

  const __bf16* krow0 = Kb + ((size_t)bh * T_ + ln) * CH_;
  const __bf16* krow1 = Kb + ((size_t)bh * T_ + 16 + ln) * CH_;
  const __bf16* vbase = Vt + ((size_t)bh * CH_ + ln) * T_;

  for (int sb = 0; sb < T_; sb += 32) {
    // --- all 8 K-block fragments first: incremental waits under the WMMAs
    v16bf kf[8];
#pragma unroll
    for (int kk = 0; kk < 4; ++kk) {
      kf[2 * kk + 0] = load_frag(krow0 + (size_t)sb * CH_ + kk * 32, koff);
      kf[2 * kk + 1] = load_frag(krow1 + (size_t)sb * CH_ + kk * 32, koff);
    }
    v8f s0 = {}, s1 = {};
#pragma unroll
    for (int kk = 0; kk < 4; ++kk) {
      s0 = wmma_bf16(qfrag[kk], kf[2 * kk + 0], s0);
      s1 = wmma_bf16(qfrag[kk], kf[2 * kk + 1], s1);
    }
    // --- issue V fragments now; latency hides under the softmax VALU below
    v16bf vf[8];
#pragma unroll
    for (int j = 0; j < 8; ++j)
      vf[j] = load_frag(vbase + (size_t)j * 16 * T_ + sb, koff);

    // online softmax: row reductions across the 16-lane half owning this row
#pragma unroll
    for (int r = 0; r < 8; ++r) {
      float mx = fmaxf(s0[r], s1[r]);
      mx = fmaxf(mx, SWZ(mx, 0x041F));
      mx = fmaxf(mx, SWZ(mx, 0x081F));
      mx = fmaxf(mx, SWZ(mx, 0x101F));
      mx = fmaxf(mx, SWZ(mx, 0x201F));
      float mnew = fmaxf(mrow[r], mx);
      float p0 = __expf(s0[r] - mnew);
      float p1 = __expf(s1[r] - mnew);
      float corr = __expf(mrow[r] - mnew);
      float rs = p0 + p1;
      rs += SWZ(rs, 0x041F);
      rs += SWZ(rs, 0x081F);
      rs += SWZ(rs, 0x101F);
      rs += SWZ(rs, 0x201F);
      lrow[r] = lrow[r] * corr + rs;
      mrow[r] = mnew;
#pragma unroll
      for (int j = 0; j < 8; ++j) oacc[j][r] = oacc[j][r] * corr;
      s0[r] = p0;
      s1[r] = p1;
    }
    // C-layout P -> LDS -> A-layout fragment (per-wave tile; same-wave DS is in-order)
#pragma unroll
    for (int r = 0; r < 8; ++r) {
      plds[w][r + hi * 8][ln] = f2bf(s0[r]);
      plds[w][r + hi * 8][16 + ln] = f2bf(s1[r]);
    }
    __builtin_amdgcn_wave_barrier();
    v16bf pfrag = load_frag(&plds[w][ln][0], koff);
    __builtin_amdgcn_wave_barrier();
    // P(16x32) . V(32x128): 8 N-tiles, V fragments already in flight/resident
#pragma unroll
    for (int j = 0; j < 8; ++j) oacc[j] = wmma_bf16(pfrag, vf[j], oacc[j]);
  }
  int b = bh >> 2, h = bh & 3;
#pragma unroll
  for (int r = 0; r < 8; ++r) {
    float inv = 1.0f / lrow[r];
    int t = qbase + r + hi * 8;
#pragma unroll
    for (int j = 0; j < 8; ++j) {
      int c = h * CH_ + j * 16 + ln;
      Ab[((size_t)b * T_ + t) * C_ + c] = f2bf(oacc[j][r] * inv);
    }
  }
}

// ---------------------------------------------------------------- out proj + residual
__global__ void k_proj(const __bf16* __restrict__ wp, const __bf16* __restrict__ Ab,
                       const float* __restrict__ bias, const float* __restrict__ x,
                       float* __restrict__ out) {
  int b = blockIdx.z;
  int w = threadIdx.x >> 5, lane = threadIdx.x & 31;
  int hi = lane >> 4, ln = lane & 15;
  int koff = hi * 8;
  int mbase = blockIdx.y * 64 + w * 16;
  int nbase = blockIdx.x * 64;
  v8f acc[4] = {};
  const __bf16* arow = wp + (size_t)(mbase + ln) * C_;
  const __bf16* bbase = Ab + ((size_t)b * T_ + nbase + ln) * C_;
  gemm_k512(acc, arow, bbase, koff);

#pragma unroll
  for (int j = 0; j < 4; ++j) {
    int t = nbase + j * 16 + ln;
#pragma unroll
    for (int r = 0; r < 8; ++r) {
      int o = mbase + r + hi * 8;
      size_t idx = ((size_t)b * C_ + o) * T_ + t;
      out[idx] = acc[j][r] + bias[o] + x[idx];
    }
  }
}

// ---------------------------------------------------------------- launcher
extern "C" void kernel_launch(void* const* d_in, const int* in_sizes, int n_in,
                              void* d_out, int out_size, void* d_ws, size_t ws_size,
                              hipStream_t stream) {
  const float* x      = (const float*)d_in[0];
  const float* qkv_w  = (const float*)d_in[1];
  const float* qkv_b  = (const float*)d_in[2];
  const float* proj_w = (const float*)d_in[3];
  const float* proj_b = (const float*)d_in[4];
  float* out = (float*)d_out;

  char* p = (char*)d_ws;
  __bf16* xT = (__bf16*)p;  p += (size_t)B_ * T_ * C_ * 2;        // 16 MB
  __bf16* wq = (__bf16*)p;  p += (size_t)O3_ * C_ * 2;            // 1.5 MB
  __bf16* wp = (__bf16*)p;  p += (size_t)C_ * C_ * 2;             // 0.5 MB
  __bf16* Q  = (__bf16*)p;  p += (size_t)B_ * H_ * T_ * CH_ * 2;  // 16 MB
  __bf16* Kb = (__bf16*)p;  p += (size_t)B_ * H_ * T_ * CH_ * 2;  // 16 MB
  __bf16* Vt = (__bf16*)p;  p += (size_t)B_ * H_ * T_ * CH_ * 2;  // 16 MB
  __bf16* Ab = (__bf16*)p;  p += (size_t)B_ * T_ * C_ * 2;        // 16 MB

  k_convert<<<(O3_ * C_ + 255) / 256, 256, 0, stream>>>(qkv_w, wq, O3_ * C_);
  k_convert<<<(C_ * C_ + 255) / 256, 256, 0, stream>>>(proj_w, wp, C_ * C_);
  k_transpose_x<<<dim3(T_ / 32, C_ / 32, B_), 256, 0, stream>>>(x, xT);
  k_qkv<<<dim3(T_ / 64, O3_ / 64, B_), 128, 0, stream>>>(wq, xT, qkv_b, Q, Kb, Vt);
  k_attn<<<dim3(T_ / 64, B_ * H_), 128, 0, stream>>>(Q, Kb, Vt, Ab);
  k_proj<<<dim3(T_ / 64, C_ / 64, B_), 128, 0, stream>>>(wp, Ab, proj_b, x, out);
}
